// RecurrentNeuralNetwork_747324310356
// MI455X (gfx1250) — matrix operand order repaired and checked by reading, not compiled, not run on hardware
//
#include <hip/hip_runtime.h>

// ---------------------------------------------------------------------------
// 2-layer LSTM for gfx1250 (MI455X):
//   - bf16 WMMA (f32 accumulate) for input projection GEMM and recurrent GEMM
//   - persistent single-WGP scan: h in LDS, c in VGPRs
//   - xp slice staging via TDM tensor_load_to_lds (TENSORcnt) in the scan loop,
//     per-lane global_load_async_to_lds_b128 (ASYNCcnt) for the prologue
// L=2, B=32, S=2048, H=256, G=4H=1024.
// ---------------------------------------------------------------------------

typedef __attribute__((ext_vector_type(16))) __bf16  v16bf;
typedef __attribute__((ext_vector_type(8)))  __bf16  v8bf;
typedef __attribute__((ext_vector_type(8)))  float   v8f;
typedef __attribute__((ext_vector_type(4)))  unsigned u32x4;
typedef __attribute__((ext_vector_type(8)))  unsigned u32x8;

#define L_ 2
#define B_ 32
#define S_ 2048
#define H_ 256
#define G_ 1024            // 4*H
#define M_ (B_ * S_)       // 65536

union FragBF {
    v16bf v;
    v8bf  h8[2];
};

__device__ __forceinline__ float sigmoidf_(float x) {
    return 1.0f / (1.0f + __expf(-x));
}

// ---- gfx1250 async copy: global -> LDS (tracked by ASYNCcnt) ---------------
__device__ __forceinline__ void async_copy_b128(unsigned lds_off, const void* gsrc) {
    asm volatile("global_load_async_to_lds_b128 %0, %1, off"
                 :: "v"(lds_off), "v"(gsrc) : "memory");
}

__device__ __forceinline__ void wait_async0() {
#if __has_builtin(__builtin_amdgcn_s_wait_asynccnt)
    __builtin_amdgcn_s_wait_asynccnt(0);
#else
    asm volatile("s_wait_asynccnt 0x0" ::: "memory");
#endif
}

__device__ __forceinline__ void wait_tensor0() {
#if __has_builtin(__builtin_amdgcn_s_wait_tensorcnt)
    __builtin_amdgcn_s_wait_tensorcnt(0);
#else
    asm volatile("s_wait_tensorcnt 0x0" ::: "memory");
#endif
}

// ---- gfx1250 Tensor Data Mover: 2D strided tile -> contiguous LDS ----------
// D# per CDNA5 ISA: group0 {count|lds_addr|global_addr|
// type=2}, group1 {data_size=2B, tensor_dim0/tile_dim0 = row_elems,
// tensor_dim1/tile_dim1 = rows, tensor_dim0_stride = row_stride_elems}.
__device__ __forceinline__ void tdm_load_2d_bf16(unsigned lds_byte_off,
                                                 const void* gsrc,
                                                 unsigned row_elems,
                                                 unsigned rows,
                                                 unsigned long long row_stride_elems) {
    const unsigned long long ga = (unsigned long long)(size_t)gsrc;
    u32x4 g0;
    g0[0] = 1u;                                            // count=1 (valid)
    g0[1] = lds_byte_off;                                  // lds_addr
    g0[2] = (unsigned)(ga & 0xFFFFFFFFu);                  // global_addr lo
    g0[3] = (unsigned)((ga >> 32) & 0x01FFFFFFu)           // global_addr hi
            | (2u << 30);                                  // type = 2 (image)
    u32x8 g1;
    g1[0] = 1u << 16;                                      // data_size = 2 B
    g1[1] = (row_elems & 0xFFFFu) << 16;                   // tensor_dim0 lo16
    g1[2] = (row_elems >> 16) | ((rows & 0xFFFFu) << 16);  // dim0 hi | dim1 lo
    g1[3] = (rows >> 16) | ((row_elems & 0xFFFFu) << 16);  // dim1 hi | tile_dim0
    g1[4] = rows & 0xFFFFu;                                // tile_dim1 (tile_dim2=0)
    g1[5] = (unsigned)(row_stride_elems & 0xFFFFFFFFu);    // dim0_stride lo32
    g1[6] = (unsigned)((row_stride_elems >> 32) & 0xFFFFu);// dim0_stride hi16
    g1[7] = 0u;                                            // dim1_stride (unused)
    asm volatile("tensor_load_to_lds %0, %1" :: "s"(g0), "s"(g1) : "memory");
}

// ---------------------------------------------------------------------------
// fp32 -> bf16 conversion (grid-stride)
// ---------------------------------------------------------------------------
__global__ void cvt_f32_bf16(const float* __restrict__ src,
                             __bf16* __restrict__ dst, int n) {
    int i      = blockIdx.x * blockDim.x + threadIdx.x;
    int stride = gridDim.x * blockDim.x;
    for (; i < n; i += stride) dst[i] = (__bf16)src[i];
}

// ---------------------------------------------------------------------------
// xp(bf16) = A(bf16 [M,H]) * Wg^T (Wg bf16 [G,H], B[k,n] = Wg[n,k]) + bias
// One 16x16 output tile per wave; K-loop of 8 x wmma_f32_16x16x32_bf16.
// ---------------------------------------------------------------------------
__global__ void __launch_bounds__(256) gemm_xp(
        const __bf16* __restrict__ A,     // [M, H] bf16 row-major
        const __bf16* __restrict__ Wg,    // [G, H] bf16 row-major
        const float*  __restrict__ bias,  // [G]
        __bf16*       __restrict__ xp)    // [M, G] bf16
{
    const int lane = threadIdx.x & 31;
    const int wave = threadIdx.x >> 5;
    const int tile = blockIdx.x * 8 + wave;   // 8 waves per block
    const int nt   = tile & (G_ / 16 - 1);    // 64 N-tiles
    const int mt   = tile / (G_ / 16);
    const int m0   = mt * 16;
    const int n0   = nt * 16;

    const int hl   = lane >> 4;   // lane-half: 0 or 1
    const int lrow = lane & 15;

    const int arow = m0 + lrow;   // A row  (M index)
    const int brow = n0 + lrow;   // Wg row (N / output column)

    v8f acc = {};
#pragma unroll
    for (int kc = 0; kc < H_ / 32; ++kc) {
        const int k0 = kc * 32;
        FragBF a, bf;
        // A 16x32 bf16 layout: lanes 0-15 K={0-7,16-23}, lanes 16-31 K={8-15,24-31}
        a.h8[0] = *(const v8bf*)(A + (size_t)arow * H_ + k0 + 8 * hl);
        a.h8[1] = *(const v8bf*)(A + (size_t)arow * H_ + k0 + 16 + 8 * hl);
        // B 32x16 bf16 layout: lanes 0-15 K={0-7,16-23}, lanes 16-31 K={8-15,24-31}
        bf.h8[0] = *(const v8bf*)(Wg + (size_t)brow * H_ + k0 + 8 * hl);
        bf.h8[1] = *(const v8bf*)(Wg + (size_t)brow * H_ + k0 + 16 + 8 * hl);
        acc = __builtin_amdgcn_wmma_f32_16x16x32_bf16(
                  false, a.v, false, bf.v, (short)0, acc, false, false);
    }

    const float bv = bias[brow];
#pragma unroll
    for (int r = 0; r < 8; ++r) {
        const int m = m0 + r + 8 * hl;        // C layout: VGPR r -> M = r (+8 for hi lanes)
        xp[(size_t)m * G_ + brow] = (__bf16)(acc[r] + bv);
    }
}

// ---------------------------------------------------------------------------
// Persistent LSTM scan for one layer: single workgroup, 32 waves (1024 thr).
// Wave w: mt = w/16 selects batch half, hh0 = 16*(w%16) the hidden columns.
// Each wave computes ALL FOUR gate tiles (i,f,g,o) for its (b,h) tile, so the
// cell update is fully in-register; c stays in VGPRs for all 2048 steps; only
// h (bf16, 16 KB) is shared through LDS. xp slices (64 KB/step, bf16) are
// staged one step ahead: wave 0 issues a single TDM tensor_load_to_lds of the
// 32x1024 strided tile; completion = s_wait_tensorcnt 0 + workgroup barrier.
// ---------------------------------------------------------------------------
__global__ void __launch_bounds__(1024, 1) lstm_scan(
        const __bf16* __restrict__ xp,   // [B*S, G] bf16 input proj (+b_ih)
        const __bf16* __restrict__ Wh,   // [G, H] bf16
        const float*  __restrict__ bh,   // [G]
        float*        __restrict__ y,    // [B, S, H] layer output
        float*        __restrict__ hT,   // [B, H]
        float*        __restrict__ cT)   // [B, H]
{
    // 16 KB h-state + 2 x 64 KB xp double buffer = 144 KB LDS
    __shared__ __bf16 smem[B_ * H_ + 2 * B_ * G_];
    __bf16* h_bf  = smem;                 // [B_, H_]
    __bf16* xbase = smem + B_ * H_;       // two [B_, G_] buffers back-to-back

    const int tid  = threadIdx.x;
    const int lane = tid & 31;
    const int wave = tid >> 5;           // 0..31
    const int mt   = wave >> 4;          // batch tile 0/1
    const int hh0  = (wave & 15) * 16;   // hidden column base
    const int hl   = lane >> 4;
    const int lrow = lane & 15;
    const int hh   = hh0 + lrow;         // this lane's hidden column
    const int arow = mt * 16 + lrow;     // this lane's A (h-state) row

    // prologue staging assignment: thread -> (batch row, 4 x 16B column chunks)
    const int prow = tid >> 5;           // 0..31 batch row
    const int pcol = (tid & 31) * 8;     // element base; chunks at +256*j

    // zero the hidden state
    for (int i = tid; i < B_ * H_; i += blockDim.x)
        h_bf[i] = (__bf16)0.0f;

    // pre-pull Wh (512 KB bf16) into L2 -> global_prefetch_b8
    for (size_t off = (size_t)tid * 64; off < (size_t)G_ * H_ * 2;
         off += (size_t)blockDim.x * 64)
        __builtin_prefetch((const char*)Wh + off, 0, 1);

    const float bh_i = bh[0 * H_ + hh];
    const float bh_f = bh[1 * H_ + hh];
    const float bh_g = bh[2 * H_ + hh];
    const float bh_o = bh[3 * H_ + hh];

    // prologue: stage step-0 xp slice with per-lane async copies (ASYNCcnt)
    {
        const __bf16* src = xp + ((size_t)prow * S_ + 0) * G_;
        const unsigned dst = (unsigned)(size_t)(xbase + prow * G_);
#pragma unroll
        for (int j = 0; j < 4; ++j)
            async_copy_b128(dst + (unsigned)(pcol + 256 * j) * 2,
                            src + pcol + 256 * j);
    }
    wait_async0();

    v8f c_reg = {};
    __syncthreads();

    for (int s = 0; s < S_; ++s) {
        const __bf16* xcur = xbase + (s & 1) * (B_ * G_);

        // wave 0 kicks off the TDM load of the next step's 32x1024 tile
        if (wave == 0 && s + 1 < S_) {
            tdm_load_2d_bf16(
                (unsigned)(size_t)(xbase + ((s + 1) & 1) * (B_ * G_)),
                xp + (size_t)(s + 1) * G_,      // tile start; rows stride S_*G_
                G_,                             // row length (elements)
                B_,                             // rows
                (unsigned long long)S_ * G_);   // row stride (elements)
        }

        v8f acc[4] = {{}, {}, {}, {}};
#pragma unroll
        for (int kc = 0; kc < H_ / 32; ++kc) {
            const int k0 = kc * 32;
            FragBF a;   // h-state fragment from LDS (ds_load_b128 x2)
            a.h8[0] = *(const v8bf*)(h_bf + arow * H_ + k0 + 8 * hl);
            a.h8[1] = *(const v8bf*)(h_bf + arow * H_ + k0 + 16 + 8 * hl);
#pragma unroll
            for (int t = 0; t < 4; ++t) {          // i, f, g, o gate tiles
                const int g = t * H_ + hh;         // Wh row for this column
                FragBF bf;
                bf.h8[0] = *(const v8bf*)(Wh + (size_t)g * H_ + k0 + 8 * hl);
                bf.h8[1] = *(const v8bf*)(Wh + (size_t)g * H_ + k0 + 16 + 8 * hl);
                acc[t] = __builtin_amdgcn_wmma_f32_16x16x32_bf16(
                             false, a.v, false, bf.v, (short)0, acc[t],
                             false, false);
            }
        }
        __syncthreads();   // all h-state reads done before anyone rewrites h

#pragma unroll
        for (int r = 0; r < 8; ++r) {
            const int    b   = mt * 16 + r + 8 * hl;
            const size_t row = (size_t)b * S_ + s;
            const float gi = acc[0][r] + (float)xcur[b * G_ + 0 * H_ + hh] + bh_i;
            const float gf = acc[1][r] + (float)xcur[b * G_ + 1 * H_ + hh] + bh_f;
            const float gg = acc[2][r] + (float)xcur[b * G_ + 2 * H_ + hh] + bh_g;
            const float go = acc[3][r] + (float)xcur[b * G_ + 3 * H_ + hh] + bh_o;

            const float iv = sigmoidf_(gi);
            const float fv = sigmoidf_(gf);
            const float gv = tanhf(gg);
            const float ov = sigmoidf_(go);

            const float c = fv * c_reg[r] + iv * gv;
            c_reg[r]      = c;
            const float h = ov * tanhf(c);

            h_bf[b * H_ + hh]  = (__bf16)h;
            y[row * H_ + hh]   = h;
            if (s == S_ - 1) {               // uniform branch
                hT[b * H_ + hh] = h;
                cT[b * H_ + hh] = c;
            }
        }

        if (wave == 0) wait_tensor0();   // TDM tile resident in LDS
        __syncthreads();                 // new h + staged xp visible to all
    }
}

// ---------------------------------------------------------------------------
// Host-side launcher
// ---------------------------------------------------------------------------
extern "C" void kernel_launch(void* const* d_in, const int* in_sizes, int n_in,
                              void* d_out, int out_size, void* d_ws, size_t ws_size,
                              hipStream_t stream) {
    (void)in_sizes; (void)n_in; (void)out_size; (void)ws_size;

    const float* x0   = (const float*)d_in[0];   // [B,S,H]
    const float* W_ih = (const float*)d_in[1];   // [L,4H,H]
    const float* b_ih = (const float*)d_in[2];   // [L,4H]
    const float* W_hh = (const float*)d_in[3];   // [L,4H,H]
    const float* b_hh = (const float*)d_in[4];   // [L,4H]

    float* out   = (float*)d_out;
    float* out_y = out;                              // [B,S,H]
    float* out_h = out + (size_t)B_ * S_ * H_;       // [L,B,H]
    float* out_c = out_h + (size_t)L_ * B_ * H_;     // [L,B,H]

    char* ws = (char*)d_ws;
    __bf16* xp   = (__bf16*)ws;                                      // 128 MB
    float*  y0   = (float*)(ws + (size_t)B_ * S_ * G_ * 2);          //  64 MB
    __bf16* xbf  = (__bf16*)((char*)y0 + (size_t)B_ * S_ * H_ * 4);  //  32 MB
    __bf16* wibf = (__bf16*)((char*)xbf + (size_t)B_ * S_ * H_ * 2); //   1 MB
    __bf16* whbf = (__bf16*)((char*)wibf + (size_t)L_ * G_ * H_ * 2);//   1 MB

    // one-time bf16 conversion of all weights (both layers)
    cvt_f32_bf16<<<2048, 256, 0, stream>>>(W_ih, wibf, L_ * G_ * H_);
    cvt_f32_bf16<<<2048, 256, 0, stream>>>(W_hh, whbf, L_ * G_ * H_);

    for (int l = 0; l < L_; ++l) {
        const float* xin  = (l == 0) ? x0 : y0;
        float*       yout = (l == L_ - 1) ? out_y : y0;

        cvt_f32_bf16<<<4096, 256, 0, stream>>>(xin, xbf, B_ * S_ * H_);

        const int n_tiles = (M_ / 16) * (G_ / 16);   // 262144 tiles, 8/block
        gemm_xp<<<n_tiles / 8, 256, 0, stream>>>(
            xbf, wibf + (size_t)l * G_ * H_, b_ih + (size_t)l * G_, xp);

        lstm_scan<<<1, 1024, 0, stream>>>(
            xp, whbf + (size_t)l * G_ * H_, b_hh + (size_t)l * G_,
            yout, out_h + (size_t)l * B_ * H_, out_c + (size_t)l * B_ * H_);
    }
}
